// GNN_36636071035404
// MI455X (gfx1250) — matrix operand order repaired and checked by reading, not compiled, not run on hardware
//
#include <hip/hip_runtime.h>
#include <hip/hip_bf16.h>

// ---------------- problem constants (match reference) ----------------
constexpr int kN = 50000;        // nodes
constexpr int kE = 800000;       // edges
constexpr int kG = 128;          // input feature dim
constexpr int kH1 = 128;         // RGCN out dim
constexpr int kHEADS = 4;
constexpr int kOUT = 128;        // H2*HEADS
constexpr int kR = 6;            // relations
constexpr float kBN_EPS = 1e-5f;
constexpr int kQKVS = 4 * kOUT;  // 512: [q|k|v|out(skip-init)] per node
constexpr int kNS = kN / 16;     // 3125 row strips of 16

typedef __attribute__((ext_vector_type(16))) _Float16 v16h;
typedef __attribute__((ext_vector_type(8)))  float    v8f;

// =====================================================================
// Fragment-order packing (ISA 7.12.2, wave32).
// A 16x32 f16 (MxK): lane -> M=lane&15;  elem i (=2v+j):
//   k = ((i>>3)<<4) + (lane&16 ? 8 : 0) + (((i>>1)&3)<<1) + (i&1)
// B 32x16 f16 (KxN): lane -> N=lane&15;  elem i: k = ((i>>3)<<4) + (lane&16 ? 8:0) + (i&7)
// Packed so one lane's 16 values are 32 contiguous bytes => 2x b128 loads.
// =====================================================================

// X [kN x kG] fp32 -> A-fragment blocks: out[((s*4+kb)*32+lane)*16+i]
__global__ void pack_a(const float* __restrict__ in, _Float16* __restrict__ out) {
  int t = blockIdx.x * blockDim.x + threadIdx.x;
  if (t >= kN * kG) return;
  int i = t & 15, lane = (t >> 4) & 31, kb = (t >> 9) & 3, s = t >> 11;
  int m = lane & 15;
  int k = kb * 32 + ((i >> 3) << 4) + ((lane & 16) ? 8 : 0) + (((i >> 1) & 3) << 1) + (i & 1);
  out[t] = (_Float16)in[(size_t)(s * 16 + m) * kG + k];
}

// W_rel [R x 128 x 128] fp32 -> B-fragment blocks per relation
__global__ void pack_wrel(const float* __restrict__ W, _Float16* __restrict__ out) {
  int t = blockIdx.x * blockDim.x + threadIdx.x;  // within one relation
  int r = blockIdx.y;
  if (t >= kG * kH1) return;
  int i = t & 15, lane = (t >> 4) & 31, kb = (t >> 9) & 3, nt = t >> 11;
  int k = kb * 32 + ((i >> 3) << 4) + ((lane & 16) ? 8 : 0) + (i & 7);
  int n = nt * 16 + (lane & 15);
  out[(size_t)r * kG * kH1 + t] = (_Float16)W[(size_t)r * kG * kH1 + k * kH1 + n];
}

// W_root [128 x 128] fp32 -> B-fragment blocks
__global__ void pack_wroot(const float* __restrict__ W, _Float16* __restrict__ out) {
  int t = blockIdx.x * blockDim.x + threadIdx.x;
  if (t >= kG * kH1) return;
  int i = t & 15, lane = (t >> 4) & 31, kb = (t >> 9) & 3, nt = t >> 11;
  int k = kb * 32 + ((i >> 3) << 4) + ((lane & 16) ? 8 : 0) + (i & 7);
  int n = nt * 16 + (lane & 15);
  out[t] = (_Float16)W[k * kH1 + n];
}

// [Wq|Wk|Wv|Wskip] -> B-fragment blocks of a 128x512 matrix, plus bcat
__global__ void pack_wcat(const float* __restrict__ Wq, const float* __restrict__ Wk,
                          const float* __restrict__ Wv, const float* __restrict__ Ws,
                          const float* __restrict__ bq, const float* __restrict__ bk,
                          const float* __restrict__ bv, const float* __restrict__ bs,
                          _Float16* __restrict__ out, float* __restrict__ bcat) {
  int t = blockIdx.x * blockDim.x + threadIdx.x;
  if (t < kH1 * kQKVS) {
    int i = t & 15, lane = (t >> 4) & 31, kb = (t >> 9) & 3, nt = t >> 11;  // nt 0..31
    int k = kb * 32 + ((i >> 3) << 4) + ((lane & 16) ? 8 : 0) + (i & 7);
    int n = nt * 16 + (lane & 15);
    int seg = n >> 7, o = n & 127;
    const float* W = (seg == 0) ? Wq : (seg == 1) ? Wk : (seg == 2) ? Wv : Ws;
    out[t] = (_Float16)W[k * kOUT + o];
  }
  if (t < kQKVS) {
    int seg = t >> 7, o = t & 127;
    const float* b = (seg == 0) ? bq : (seg == 1) ? bk : (seg == 2) ? bv : bs;
    bcat[t] = b[o];
  }
}

// =====================================================================
// GEMM 1: XT[r] = X @ W_rel[r], f16 out.
// One wave per 16-row strip; A fragments (K=128) cached in registers and
// reused across R*8 = 48 output tiles. All fragment loads are 2x b128.
// =====================================================================
__global__ void gemm_xt(const _Float16* __restrict__ Ab, const _Float16* __restrict__ Bb,
                        _Float16* __restrict__ XT) {
  int wave = threadIdx.x >> 5, lane = threadIdx.x & 31;
  int strip = blockIdx.x * 8 + wave;
  if (strip >= kNS) return;
  const v16h* A = (const v16h*)(Ab + (size_t)strip * (4 * 32 * 16));
  v16h Af[4];
#pragma unroll
  for (int kb = 0; kb < 4; ++kb) Af[kb] = A[kb * 32 + lane];
  int n = lane & 15;
  int mhi = (lane & 16) ? 8 : 0;
  for (int r = 0; r < kR; ++r) {
    const v16h* B = (const v16h*)(Bb + (size_t)r * kG * kH1);
    _Float16* C = XT + ((size_t)r * kN + (size_t)strip * 16) * kH1;
#pragma unroll
    for (int nt = 0; nt < 8; ++nt) {
      v8f acc = {};
#pragma unroll
      for (int kb = 0; kb < 4; ++kb)
        acc = __builtin_amdgcn_wmma_f32_16x16x32_f16(false, Af[kb], false,
                                                     B[(nt * 4 + kb) * 32 + lane],
                                                     (short)0, acc, false, false);
      _Float16* Cc = C + nt * 16 + n;
#pragma unroll
      for (int v = 0; v < 8; ++v)
        Cc[(size_t)(v + mhi) * kH1] = (_Float16)acc[v];
    }
  }
}

// ---------------- RGCN edge passes ----------------
__global__ void edge_count(const int* __restrict__ eidx, const int* __restrict__ etype,
                           float* __restrict__ cnt) {
  int e = blockIdx.x * blockDim.x + threadIdx.x;
  if (e >= kE) return;
  atomicAdd(&cnt[(size_t)etype[e] * kN + eidx[kE + e]], 1.0f);
}

// agg[dst,f] += XT[r,src,f]/cnt[r,dst]   (mean per (r,dst), summed over r)
__global__ void edge_agg(const int* __restrict__ eidx, const int* __restrict__ etype,
                         const _Float16* __restrict__ XT, const float* __restrict__ cnt,
                         float* __restrict__ agg) {
  long long t = (long long)blockIdx.x * blockDim.x + threadIdx.x;
  int e = (int)(t >> 7);
  if (e >= kE) return;
  int f = (int)(t & 127);
  int r = etype[e], s = eidx[e], d = eidx[kE + e];
  float c = cnt[(size_t)r * kN + d];  // >= 1: this edge exists in its segment
  float v = (float)XT[((size_t)r * kN + s) * kH1 + f] / c;
  atomicAdd(&agg[(size_t)d * kH1 + f], v);
}

// =====================================================================
// GEMM 2: x = X @ W_root + b + agg.  Epilogue writes x DIRECTLY in
// A-fragment layout (xb2) for the downstream QKVS GEMM — no re-swizzle pass.
// (m,k)->(lane,elem) inverse: lane2 = m | ((k&8)?16:0),
//                             i2 = (k&7) | ((k&16)?8:0)
// =====================================================================
__global__ void gemm_x(const _Float16* __restrict__ Ab, const _Float16* __restrict__ Bb,
                       const float* __restrict__ brg, const float* __restrict__ agg,
                       _Float16* __restrict__ xb2) {
  int wave = threadIdx.x >> 5, lane = threadIdx.x & 31;
  int strip = blockIdx.x * 8 + wave;
  if (strip >= kNS) return;
  const v16h* A = (const v16h*)(Ab + (size_t)strip * (4 * 32 * 16));
  const v16h* B = (const v16h*)Bb;
  v16h Af[4];
#pragma unroll
  for (int kb = 0; kb < 4; ++kb) Af[kb] = A[kb * 32 + lane];
  int n = lane & 15;
  int mhi = (lane & 16) ? 8 : 0;
#pragma unroll
  for (int nt = 0; nt < 8; ++nt) {
    v8f acc = {};
#pragma unroll
    for (int kb = 0; kb < 4; ++kb)
      acc = __builtin_amdgcn_wmma_f32_16x16x32_f16(false, Af[kb], false,
                                                   B[(nt * 4 + kb) * 32 + lane],
                                                   (short)0, acc, false, false);
#pragma unroll
    for (int v = 0; v < 8; ++v) {
      int m = v + mhi;                       // row within strip
      int col = nt * 16 + n;                 // feature (= k of next GEMM)
      float val = acc[v] + brg[col] + agg[(size_t)(strip * 16 + m) * kH1 + col];
      int kb2 = col >> 5, kp = col & 31;
      int lane2 = m | ((kp & 8) ? 16 : 0);
      int i2 = (kp & 7) | ((kp & 16) ? 8 : 0);
      xb2[(((size_t)strip * 4 + kb2) * 32 + lane2) * 16 + i2] = (_Float16)val;
    }
  }
}

// =====================================================================
// GEMM 3: QKVS = x @ [Wq|Wk|Wv|Wskip] + bcat, fp32 out [kN x 512].
// Wave per strip, A fragments cached, 32 column tiles.
// =====================================================================
__global__ void gemm_qkvs(const _Float16* __restrict__ xb2, const _Float16* __restrict__ Bb,
                          const float* __restrict__ bcat, float* __restrict__ QKVS) {
  int wave = threadIdx.x >> 5, lane = threadIdx.x & 31;
  int strip = blockIdx.x * 8 + wave;
  if (strip >= kNS) return;
  const v16h* A = (const v16h*)(xb2 + (size_t)strip * (4 * 32 * 16));
  const v16h* B = (const v16h*)Bb;
  v16h Af[4];
#pragma unroll
  for (int kb = 0; kb < 4; ++kb) Af[kb] = A[kb * 32 + lane];
  int n = lane & 15;
  int mhi = (lane & 16) ? 8 : 0;
#pragma unroll 4
  for (int nt = 0; nt < 32; ++nt) {
    v8f acc = {};
#pragma unroll
    for (int kb = 0; kb < 4; ++kb)
      acc = __builtin_amdgcn_wmma_f32_16x16x32_f16(false, Af[kb], false,
                                                   B[(nt * 4 + kb) * 32 + lane],
                                                   (short)0, acc, false, false);
    int col = nt * 16 + n;
    float bc = bcat[col];
#pragma unroll
    for (int v = 0; v < 8; ++v)
      QKVS[(size_t)(strip * 16 + v + mhi) * kQKVS + col] = acc[v] + bc;
  }
}

// ---------------- attention (segment softmax over dst) ----------------
__device__ inline unsigned fkey(float f) {
  unsigned b = __float_as_uint(f);
  return (b & 0x80000000u) ? ~b : (b | 0x80000000u);  // monotone float->uint
}
__device__ inline float funkey(unsigned u) {
  unsigned b = (u & 0x80000000u) ? (u & 0x7fffffffu) : ~u;
  return __uint_as_float(b);
}

__global__ void attn_score_max(const int* __restrict__ eidx, const float* __restrict__ QKVS,
                               float* __restrict__ score, unsigned* __restrict__ mkey) {
  int t = blockIdx.x * blockDim.x + threadIdx.x;  // over E*HEADS
  if (t >= kE * kHEADS) return;
  int e = t >> 2, h = t & 3;
  int s = eidx[e], d = eidx[kE + e];
  const float4* q4 = (const float4*)(QKVS + (size_t)d * kQKVS + h * 32);
  const float4* k4 = (const float4*)(QKVS + (size_t)s * kQKVS + kOUT + h * 32);
  float sc = 0.f;
#pragma unroll
  for (int c = 0; c < 8; ++c) {
    float4 a = q4[c], b = k4[c];
    sc += a.x * b.x + a.y * b.y + a.z * b.z + a.w * b.w;
  }
  sc *= 0.17677669529663687f;  // 1/sqrt(H2)
  score[t] = sc;
  atomicMax(&mkey[(size_t)d * kHEADS + h], fkey(sc));
}

__global__ void decode_m(const unsigned* __restrict__ mkey, float* __restrict__ mval) {
  int t = blockIdx.x * blockDim.x + threadIdx.x;
  if (t >= kN * kHEADS) return;
  unsigned u = mkey[t];
  mval[t] = (u == 0u) ? 0.f : funkey(u);  // isolated node -> m=0 (matches ref)
}

__global__ void attn_den(const int* __restrict__ eidx, const float* __restrict__ score,
                         const float* __restrict__ mval, float* __restrict__ ex,
                         float* __restrict__ den) {
  int t = blockIdx.x * blockDim.x + threadIdx.x;
  if (t >= kE * kHEADS) return;
  int e = t >> 2, h = t & 3;
  int d = eidx[kE + e];
  float x = __expf(score[t] - mval[(size_t)d * kHEADS + h]);
  ex[t] = x;
  atomicAdd(&den[(size_t)d * kHEADS + h], x);
}

// out(cols 384:512 of QKVS, pre-initialized with skip) += alpha * v[src]
__global__ void attn_out(const int* __restrict__ eidx, const float* __restrict__ ex,
                         const float* __restrict__ den, float* __restrict__ QKVS) {
  long long t = (long long)blockIdx.x * blockDim.x + threadIdx.x;  // over E*128
  int e = (int)(t >> 7);
  if (e >= kE) return;
  int f = (int)(t & 127), h = f >> 5;
  int s = eidx[e], d = eidx[kE + e];
  float alpha = ex[(size_t)e * kHEADS + h] / (den[(size_t)d * kHEADS + h] + 1e-16f);
  atomicAdd(&QKVS[(size_t)d * kQKVS + 3 * kOUT + f],
            alpha * QKVS[(size_t)s * kQKVS + 2 * kOUT + f]);
}

// ---------------- BatchNorm(train stats) + LeakyReLU ----------------
__global__ void bn_stats(const float* __restrict__ QKVS, float* __restrict__ bnsum,
                         float* __restrict__ bnsum2) {
  int c = threadIdx.x;  // 128 threads = 128 columns
  float s = 0.f, s2 = 0.f;
  for (int row = blockIdx.x; row < kN; row += gridDim.x) {
    float v = QKVS[(size_t)row * kQKVS + 3 * kOUT + c];
    s += v;
    s2 += v * v;
  }
  atomicAdd(&bnsum[c], s);
  atomicAdd(&bnsum2[c], s2);
}

__global__ void bn_apply(const float* __restrict__ QKVS, const float* __restrict__ bnsum,
                         const float* __restrict__ bnsum2, const float* __restrict__ gamma,
                         const float* __restrict__ beta, float* __restrict__ out) {
  int t = blockIdx.x * blockDim.x + threadIdx.x;  // over N*128
  if (t >= kN * kOUT) return;
  int row = t >> 7, c = t & 127;
  float inv_n = 1.0f / (float)kN;
  float mu = bnsum[c] * inv_n;
  float var = bnsum2[c] * inv_n - mu * mu;
  float v = QKVS[(size_t)row * kQKVS + 3 * kOUT + c];
  float y = (v - mu) * rsqrtf(var + kBN_EPS) * gamma[c] + beta[c];
  out[t] = (y > 0.f) ? y : 0.01f * y;
}

// ---------------- host-side orchestration ----------------
extern "C" void kernel_launch(void* const* d_in, const int* in_sizes, int n_in,
                              void* d_out, int out_size, void* d_ws, size_t ws_size,
                              hipStream_t stream) {
  (void)in_sizes; (void)n_in; (void)out_size; (void)ws_size;
  const float* nodef  = (const float*)d_in[0];
  // d_in[1] = node_type (unused, matches reference)
  const int*   eidx   = (const int*)d_in[2];
  const int*   etype  = (const int*)d_in[3];
  const float* W_rel  = (const float*)d_in[4];
  const float* W_root = (const float*)d_in[5];
  const float* b_rgcn = (const float*)d_in[6];
  const float* Wq = (const float*)d_in[7],  *bq = (const float*)d_in[8];
  const float* Wk = (const float*)d_in[9],  *bk = (const float*)d_in[10];
  const float* Wv = (const float*)d_in[11], *bv = (const float*)d_in[12];
  const float* Wsk = (const float*)d_in[13], *bsk = (const float*)d_in[14];
  const float* gamma = (const float*)d_in[15];
  const float* beta  = (const float*)d_in[16];
  float* out = (float*)d_out;

  // ---- carve workspace ----
  char* ws = (char*)d_ws;
  auto carve = [&](size_t bytes) {
    char* p = ws;
    ws += (bytes + 255) & ~(size_t)255;
    return p;
  };
  _Float16* Ab     = (_Float16*)carve((size_t)kN * kG * 2);        // X, fragment layout
  _Float16* Wrelb  = (_Float16*)carve((size_t)kR * kG * kH1 * 2);  // B fragment layout
  _Float16* Wrootb = (_Float16*)carve((size_t)kG * kH1 * 2);
  _Float16* Wcatb  = (_Float16*)carve((size_t)kH1 * kQKVS * 2);
  float*    bcat   = (float*)carve((size_t)kQKVS * 4);
  _Float16* XT     = (_Float16*)carve((size_t)kR * kN * kH1 * 2);  // row-major f16
  _Float16* xb2    = (_Float16*)carve((size_t)kN * kH1 * 2);       // x, fragment layout
  float*    QKVS   = (float*)carve((size_t)kN * kQKVS * 4);
  float*    score  = (float*)carve((size_t)kE * kHEADS * 4);
  float*    ex     = (float*)carve((size_t)kE * kHEADS * 4);
  float*    mval   = (float*)carve((size_t)kN * kHEADS * 4);
  // contiguous zero-init region A: cnt + agg
  size_t zAbytes = (size_t)kR * kN * 4 + (size_t)kN * kH1 * 4;
  char* zA = carve(zAbytes);
  float* cnt = (float*)zA;
  float* agg = (float*)(zA + (size_t)kR * kN * 4);
  // contiguous zero-init region B: mkey + den + bnsum + bnsum2
  size_t zBbytes = (size_t)kN * kHEADS * 4 * 2 + 2 * kOUT * 4;
  char* zB = carve(zBbytes);
  unsigned* mkey = (unsigned*)zB;
  float* den    = (float*)(zB + (size_t)kN * kHEADS * 4);
  float* bnsum  = (float*)(zB + (size_t)kN * kHEADS * 8);
  float* bnsum2 = bnsum + kOUT;

  hipMemsetAsync(zA, 0, zAbytes, stream);
  hipMemsetAsync(zB, 0, zBbytes, stream);

  // ---- pack operands into WMMA fragment order (one-time swizzle) ----
  pack_a<<<(kN * kG + 255) / 256, 256, 0, stream>>>(nodef, Ab);
  pack_wrel<<<dim3((kG * kH1 + 255) / 256, kR), 256, 0, stream>>>(W_rel, Wrelb);
  pack_wroot<<<(kG * kH1 + 255) / 256, 256, 0, stream>>>(W_root, Wrootb);
  pack_wcat<<<(kH1 * kQKVS + 255) / 256, 256, 0, stream>>>(Wq, Wk, Wv, Wsk, bq, bk, bv, bsk,
                                                           Wcatb, bcat);

  int gemm_blocks = (kNS + 7) / 8;  // 8 waves/block, one strip per wave

  // ---- RGCN ----
  gemm_xt<<<gemm_blocks, 256, 0, stream>>>(Ab, Wrelb, XT);
  edge_count<<<(kE + 255) / 256, 256, 0, stream>>>(eidx, etype, cnt);
  {
    long long work = (long long)kE * kH1;
    edge_agg<<<(unsigned)((work + 255) / 256), 256, 0, stream>>>(eidx, etype, XT, cnt, agg);
  }
  gemm_x<<<gemm_blocks, 256, 0, stream>>>(Ab, Wrootb, b_rgcn, agg, xb2);

  // ---- TransformerConv ----
  gemm_qkvs<<<gemm_blocks, 256, 0, stream>>>(xb2, Wcatb, bcat, QKVS);
  attn_score_max<<<(kE * kHEADS + 255) / 256, 256, 0, stream>>>(eidx, QKVS, score, mkey);
  decode_m<<<(kN * kHEADS + 255) / 256, 256, 0, stream>>>(mkey, mval);
  attn_den<<<(kE * kHEADS + 255) / 256, 256, 0, stream>>>(eidx, score, mval, ex, den);
  {
    long long work = (long long)kE * kOUT;
    attn_out<<<(unsigned)((work + 255) / 256), 256, 0, stream>>>(eidx, ex, den, QKVS);
  }

  // ---- BatchNorm + LeakyReLU ----
  bn_stats<<<512, 128, 0, stream>>>(QKVS, bnsum, bnsum2);
  bn_apply<<<(kN * kOUT + 255) / 256, 256, 0, stream>>>(QKVS, bnsum, bnsum2, gamma, beta, out);
}
